// GNNBlock_26268019982879
// MI455X (gfx1250) — matrix operand order repaired and checked by reading, not compiled, not run on hardware
//
#include <hip/hip_runtime.h>

#define DIM 256

typedef __attribute__((ext_vector_type(16))) _Float16 v16h;
typedef __attribute__((ext_vector_type(8)))  _Float16 v8h;
typedef __attribute__((ext_vector_type(8)))  float    v8f;
typedef __attribute__((ext_vector_type(4)))  float    v4f;

// ---------- degree / normalization ----------
__global__ void k_init_deg(float* __restrict__ deg, int n) {
  int i = blockIdx.x * blockDim.x + threadIdx.x;
  if (i < n) deg[i] = 1.0f;  // self-loop
}

__global__ void k_count_deg(const long long* __restrict__ dst,
                            float* __restrict__ deg, int E) {
  int e = blockIdx.x * blockDim.x + threadIdx.x;
  if (e < E) unsafeAtomicAdd(&deg[(int)dst[e]], 1.0f);
}

__global__ void k_dinv(float* __restrict__ deg, int n) {
  int i = blockIdx.x * blockDim.x + threadIdx.x;
  if (i < n) deg[i] = rsqrtf(deg[i]);  // deg >= 1 always
}

// ---------- W -> f16 transposed (Wt[n][k] = W[k][n]) ----------
__global__ void k_wt16(const float* __restrict__ W, _Float16* __restrict__ Wt) {
  int idx = blockIdx.x * blockDim.x + threadIdx.x;  // 0..65535
  int ncol = idx >> 8;
  int k    = idx & 255;
  Wt[idx] = (_Float16)W[k * DIM + ncol];
}

// ---------- h = x @ W via v_wmma_f32_16x16x32_f16 ----------
__global__ __launch_bounds__(256) void k_gemm(const float* __restrict__ x,
                                              const _Float16* __restrict__ Wt,
                                              float* __restrict__ h, int n) {
  const int lane = threadIdx.x & 31;
  const int wave = threadIdx.x >> 5;        // 0..7, owns cols [32*wave, 32*wave+32)
  const int m    = lane & 15;
  const int hlf  = lane >> 4;
  const int row0 = blockIdx.x * 16;
  int arow = row0 + m; if (arow > n - 1) arow = n - 1;   // clamp (n % 16 == 0 anyway)
  const int colbase = wave * 32;

  const float*    xrow = x  + (size_t)arow * DIM;
  const _Float16* b0p  = Wt + (size_t)(colbase +      (lane & 15)) * DIM;
  const _Float16* b1p  = Wt + (size_t)(colbase + 16 + (lane & 15)) * DIM;

  v8f acc0 = {}; v8f acc1 = {};

  for (int kk = 0; kk < DIM; kk += 32) {
    // A fragment: two contiguous 8-float runs, convert f32 -> f16
    int ka = kk + 8 * hlf;
    v4f f0 = *(const v4f*)(xrow + ka);
    v4f f1 = *(const v4f*)(xrow + ka + 4);
    v4f f2 = *(const v4f*)(xrow + ka + 16);
    v4f f3 = *(const v4f*)(xrow + ka + 20);
    v8h lo, hi;
#pragma unroll
    for (int i = 0; i < 4; ++i) {
      lo[i]     = (_Float16)f0[i];
      lo[i + 4] = (_Float16)f1[i];
      hi[i]     = (_Float16)f2[i];
      hi[i + 4] = (_Float16)f3[i];
    }
    v16h a = __builtin_shufflevector(lo, hi, 0,1,2,3,4,5,6,7,8,9,10,11,12,13,14,15);

    // B fragments: two 8-half runs (k = 8h.., 16+8h..) from transposed W
    int kb = kk + 8 * hlf;
    v8h bl0 = *(const v8h*)(b0p + kb);
    v8h bh0 = *(const v8h*)(b0p + kb + 16);
    v16h b0 = __builtin_shufflevector(bl0, bh0, 0,1,2,3,4,5,6,7,8,9,10,11,12,13,14,15);
    acc0 = __builtin_amdgcn_wmma_f32_16x16x32_f16(false, a, false, b0,
                                                  (short)0, acc0, false, false);

    v8h bl1 = *(const v8h*)(b1p + kb);
    v8h bh1 = *(const v8h*)(b1p + kb + 16);
    v16h b1 = __builtin_shufflevector(bl1, bh1, 0,1,2,3,4,5,6,7,8,9,10,11,12,13,14,15);
    acc1 = __builtin_amdgcn_wmma_f32_16x16x32_f16(false, a, false, b1,
                                                  (short)0, acc1, false, false);
  }

  // D layout: element j -> row j + 8*hlf, col lane&15
  const int nl = lane & 15;
#pragma unroll
  for (int j = 0; j < 8; ++j) {
    int r = row0 + j + 8 * hlf;
    if (r < n) {
      h[(size_t)r * DIM + colbase + nl]      = acc0[j];
      h[(size_t)r * DIM + colbase + 16 + nl] = acc1[j];
    }
  }
}

// ---------- out[v][:] = h[v][:] * dinv[v]^2  (self-loop message) ----------
__global__ void k_selfinit(const float* __restrict__ h, const float* __restrict__ dinv,
                           float* __restrict__ out, int total) {
  int i = blockIdx.x * blockDim.x + threadIdx.x;
  if (i < total) {
    float di = dinv[i >> 8];
    out[i] = h[i] * di * di;
  }
}

// ---------- edge scatter: one wave per edge, lane owns 8 dims ----------
__global__ __launch_bounds__(256) void k_scatter(const long long* __restrict__ src,
                                                 const long long* __restrict__ dst,
                                                 const float* __restrict__ h,
                                                 const float* __restrict__ dinv,
                                                 float* __restrict__ out, int E) {
  int e = blockIdx.x * 8 + (threadIdx.x >> 5);
  if (e >= E) return;
  int lane = threadIdx.x & 31;
  int s = (int)src[e];
  int d = (int)dst[e];
  float norm = dinv[s] * dinv[d];
  const float* hp = h   + (size_t)s * DIM + lane * 8;
  float*       op = out + (size_t)d * DIM + lane * 8;
  v4f a0 = *(const v4f*)(hp);
  v4f a1 = *(const v4f*)(hp + 4);
#pragma unroll
  for (int i = 0; i < 4; ++i) {
    unsafeAtomicAdd(op + i,     a0[i] * norm);
    unsafeAtomicAdd(op + 4 + i, a1[i] * norm);
  }
}

// ---------- out = relu(out + b) + x ----------
__global__ void k_final(float* __restrict__ out, const float* __restrict__ x,
                        const float* __restrict__ b, int total) {
  int i = blockIdx.x * blockDim.x + threadIdx.x;
  if (i < total) {
    float v = out[i] + b[i & 255];
    out[i] = fmaxf(v, 0.0f) + x[i];
  }
}

extern "C" void kernel_launch(void* const* d_in, const int* in_sizes, int n_in,
                              void* d_out, int out_size, void* d_ws, size_t ws_size,
                              hipStream_t stream) {
  const float*     x  = (const float*)d_in[0];
  const long long* ei = (const long long*)d_in[1];
  const float*     W  = (const float*)d_in[2];
  const float*     b  = (const float*)d_in[3];
  float*           out = (float*)d_out;

  const int n = in_sizes[0] / DIM;   // 50000
  const int E = in_sizes[1] / 2;     // 800000
  const long long* src = ei;
  const long long* dst = ei + E;

  // workspace layout (256B-aligned slices)
  char* ws = (char*)d_ws;
  size_t off = 0;
  float* deg = (float*)(ws + off);                 // n floats; becomes dinv in place
  off += ((size_t)n * sizeof(float) + 255) & ~(size_t)255;
  _Float16* Wt = (_Float16*)(ws + off);            // DIM*DIM halves (128 KB)
  off += ((size_t)DIM * DIM * sizeof(_Float16) + 255) & ~(size_t)255;
  float* h = (float*)(ws + off);                   // n*DIM floats (51.2 MB)

  const int total = n * DIM;

  k_init_deg <<<(n + 255) / 256, 256, 0, stream>>>(deg, n);
  k_count_deg<<<(E + 255) / 256, 256, 0, stream>>>(dst, deg, E);
  k_dinv     <<<(n + 255) / 256, 256, 0, stream>>>(deg, n);
  k_wt16     <<<(DIM * DIM) / 256, 256, 0, stream>>>(W, Wt);
  k_gemm     <<<(n + 15) / 16, 256, 0, stream>>>(x, Wt, h, n);
  k_selfinit <<<(total + 255) / 256, 256, 0, stream>>>(h, deg, out, total);
  k_scatter  <<<(E + 7) / 8, 256, 0, stream>>>(src, dst, h, deg, out, E);
  k_final    <<<(total + 255) / 256, 256, 0, stream>>>(out, x, b, total);
}
